// mobilnet_roifeatures_crossattention_83141976916047
// MI455X (gfx1250) — matrix operand order repaired and checked by reading, not compiled, not run on hardware
//
#include <hip/hip_runtime.h>
#include <math.h>

// ---------------------------------------------------------------------------
// Types for CDNA5 WMMA (wave32)
// ---------------------------------------------------------------------------
typedef __bf16 bf16;
typedef __attribute__((ext_vector_type(16))) __bf16 v16bf;
typedef __attribute__((ext_vector_type(8)))  float  v8f;

#define B_    128
#define R_    99
#define H_    1024
#define V_    10000
#define T_    64
#define NREG  100      // 1 image feature + 99 roi features
#define G_    4096     // 4*H gates
#define K2_   2048     // 2*H concat
#define VP_   10016    // V padded to multiple of 16
#define SOS_  1
#define MT_   8        // M tiles per wave (M panel = 128 rows)

__device__ __forceinline__ float sigm_(float x) { return 1.0f / (1.0f + __expf(-x)); }

// ---------------------------------------------------------------------------
// WMMA fragment layouts (CDNA5 ISA 7.12.2, 16-bit data, wave32)
// A: 16x32 (MxK) row-major in memory. lane m = L&15, half h = L>>4,
//   slot v in 0..3: K = 2v + 8h ; slot v in 4..7: K = 16 + 2(v-4) + 8h
// B: 32x16 (KxN) PRE-PACKED: each (kTile,nTile) fragment is one contiguous
//   1KB block laid out [h 0..1][n 0..15][k-slot 0..15]; blocks ordered [nTile][kTile].
//   Lane L loads its row at base + L*32B -> fully coalesced, and a wave
//   streams B sequentially (+1KB per k-step).
// C/D: lane L -> n = L&15 ; acc[r] -> m = r + 8*(L>>4)
// ---------------------------------------------------------------------------
__device__ __forceinline__ v16bf load_a_frag(const bf16* __restrict__ A, int lda, int lane) {
    union { v16bf v; unsigned int u[8]; } f;
    const int m = lane & 15;
    const int h = lane >> 4;
    const bf16* row = A + (long)m * lda;
#pragma unroll
    for (int v = 0; v < 8; ++v) {
        const int k = (v < 4) ? (2 * v + 8 * h) : (16 + 2 * (v - 4) + 8 * h);
        f.u[v] = *(const unsigned int*)(row + k);   // two packed bf16
    }
    return f.v;
}

// ---------------------------------------------------------------------------
// WMMA GEMM, weight-streaming-optimized for small M:
//   C[M,N] = act( A0[M,K] @ B0[K,N] (+ A1 @ B1) + bias ),  B fragment-packed.
// One wave owns a 16-wide N strip and accumulates MT_=8 M-tiles (128 rows):
// each weight fragment is loaded ONCE and reused 8x; 8 independent
// accumulator chains remove WMMA->WMMA RAW hazard NOPs.
// MODE 0: dual A/B (LSTM gates), f32 out        MODE 1: tanh, bf16 out
// MODE 2: logits -> f32 scratch + d_out strided by T, guard n < V_
// Grid: x = ceil(N/64) blocks of 4 waves ; y = M/128 panels.
// ---------------------------------------------------------------------------
template <int MODE>
__global__ __launch_bounds__(128) void wmma_gemm_k(
    const bf16* __restrict__ A0, const bf16* __restrict__ Bp0,
    const bf16* __restrict__ A1, const bf16* __restrict__ Bp1,
    const float* __restrict__ bias,
    float* __restrict__ Cf, bf16* __restrict__ Cb, float* __restrict__ out2,
    int M, int N, int K, int lda, int ldc)
{
    const int lane  = threadIdx.x & 31;
    const int wave  = threadIdx.x >> 5;
    const int nTile = blockIdx.x * 4 + wave;
    const int n0 = nTile * 16;
    const int mPan = blockIdx.y * (MT_ * 16);       // 128-row M panel origin
    if (n0 >= N) return;

    const int KT = K >> 5;                          // k-tiles
    const bf16* b0p = Bp0 + (((long)nTile * KT) << 9) + lane * 16;
    const bf16* b1p = (MODE == 0) ? (Bp1 + (((long)nTile * KT) << 9) + lane * 16) : nullptr;

    v8f acc[MT_];
#pragma unroll
    for (int mt = 0; mt < MT_; ++mt) acc[mt] = v8f{};

    for (int k = 0; k < K; k += 32) {
        const v16bf b0 = *(const v16bf*)b0p;        // 32B/lane, coalesced
        b0p += 512;
        if (MODE == 0) {
            const v16bf b1 = *(const v16bf*)b1p;
            b1p += 512;
#pragma unroll
            for (int mt = 0; mt < MT_; ++mt) {
                const long mrow = (long)(mPan + mt * 16);
                v16bf a0 = load_a_frag(A0 + mrow * lda + k, lda, lane);
                acc[mt] = __builtin_amdgcn_wmma_f32_16x16x32_bf16(
                    false, a0, false, b0, (short)0, acc[mt], false, false);
                v16bf a1 = load_a_frag(A1 + mrow * lda + k, lda, lane);
                acc[mt] = __builtin_amdgcn_wmma_f32_16x16x32_bf16(
                    false, a1, false, b1, (short)0, acc[mt], false, false);
            }
        } else {
#pragma unroll
            for (int mt = 0; mt < MT_; ++mt) {
                const long mrow = (long)(mPan + mt * 16);
                v16bf a0 = load_a_frag(A0 + mrow * lda + k, lda, lane);
                acc[mt] = __builtin_amdgcn_wmma_f32_16x16x32_bf16(
                    false, a0, false, b0, (short)0, acc[mt], false, false);
            }
        }
    }

    const int h = lane >> 4;
    const int n = n0 + (lane & 15);
    float bv = 0.0f;
    if (MODE == 2) { if (n < V_) bv = bias[n]; }
    else           { bv = bias[n]; }

#pragma unroll
    for (int mt = 0; mt < MT_; ++mt) {
#pragma unroll
        for (int r = 0; r < 8; ++r) {
            const int m = mPan + mt * 16 + r + 8 * h;
            const float x = acc[mt][r] + bv;
            if (MODE == 0) {
                Cf[(long)m * ldc + n] = x;
            } else if (MODE == 1) {
                Cb[(long)m * ldc + n] = (bf16)tanhf(x);
            } else {
                if (n < V_) {
                    Cf[(long)m * V_ + n] = x;                // scratch for argmax
                    out2[((long)m * V_ + n) * T_] = x;       // d_out[b, n, t]
                }
            }
        }
    }
}

// ---------------------------------------------------------------------------
// One-time helpers
// ---------------------------------------------------------------------------
// W[N,K] fp32 -> fragment-packed bf16 B (zero-pad n >= N):
//   Wp[ (nTile*(K/32)+kTile)*512 + ((krow>>3)&1)*256 + nl*16 + (krow&7) + (krow>>4)*8 ] = W[n,k]
__global__ void transpose_pack_k(const float* __restrict__ W, bf16* __restrict__ Wp,
                                 int N, int K, int Nout)
{
    const int idx = blockIdx.x * 256 + threadIdx.x;
    const int k = idx / Nout, n = idx % Nout;
    if (k >= K) return;
    const float v = (n < N) ? W[(long)n * K + k] : 0.0f;
    const int kt = k >> 5, kr = k & 31, nt = n >> 4, nl = n & 15;
    Wp[(((long)nt * (K >> 5) + kt) << 9) + ((kr >> 3) & 1) * 256 + nl * 16 + (kr & 7) + ((kr >> 4) << 3)] = (bf16)v;
}

__global__ void add_bias_k(const float* __restrict__ a, const float* __restrict__ b,
                           float* __restrict__ o, int n)
{
    const int i = blockIdx.x * 256 + threadIdx.x;
    if (i < n) o[i] = a[i] + b[i];
}

// imgf[b] = img_feat[b] @ projcnn_w.T + b ; write as region 0 of roi_bf16; mask[b,0]
__global__ __launch_bounds__(256) void proj_img_k(
    const float* __restrict__ img, const float* __restrict__ pw,
    const float* __restrict__ pb, bf16* __restrict__ roi_b, float* __restrict__ mask)
{
    const int b = blockIdx.x, tid = threadIdx.x;
    __shared__ float sh[768];
    __shared__ float red[256];
    for (int d = tid; d < 768; d += 256) sh[d] = img[b * 768 + d];
    __syncthreads();
    float psum = 0.0f;
    for (int e = tid; e < H_; e += 256) {
        const float* wr = pw + (long)e * 768;
        float s = pb[e];
        for (int d = 0; d < 768; ++d) s += sh[d] * wr[d];
        roi_b[((long)b * NREG) * H_ + e] = (bf16)s;
        psum += s;
    }
    red[tid] = psum; __syncthreads();
    for (int s = 128; s > 0; s >>= 1) { if (tid < s) red[tid] += red[tid + s]; __syncthreads(); }
    if (tid == 0) mask[b * NREG] = (red[0] == 0.0f) ? -INFINITY : 0.0f;
}

// copy roi_features[b,n] -> roi_bf16[b,n+1], mask[b,n+1]
__global__ __launch_bounds__(256) void prep_roi_k(
    const float* __restrict__ roi, bf16* __restrict__ roi_b, float* __restrict__ mask)
{
    const int bn = blockIdx.x, b = bn / R_, n = bn % R_, tid = threadIdx.x;
    __shared__ float red[256];
    const float* src = roi + ((long)b * R_ + n) * H_;
    bf16* dst = roi_b + ((long)b * NREG + (n + 1)) * H_;
    float psum = 0.0f;
    for (int e = tid; e < H_; e += 256) { const float v = src[e]; dst[e] = (bf16)v; psum += v; }
    red[tid] = psum; __syncthreads();
    for (int s = 128; s > 0; s >>= 1) { if (tid < s) red[tid] += red[tid + s]; __syncthreads(); }
    if (tid == 0) mask[b * NREG + n + 1] = (red[0] == 0.0f) ? -INFINITY : 0.0f;
}

__global__ void set_start_k(const float* __restrict__ embed,
                            bf16* __restrict__ xb, bf16* __restrict__ xcat)
{
    const int b = blockIdx.x, tid = threadIdx.x;
    for (int e = tid; e < H_; e += 256) {
        const bf16 v = (bf16)embed[SOS_ * H_ + e];
        xb[(long)b * H_ + e] = v;
        xcat[(long)b * K2_ + H_ + e] = v;
    }
}

__global__ void init_state_k(const float* __restrict__ ctx, bf16* __restrict__ h0,
                             bf16* __restrict__ h1, float* __restrict__ c0,
                             float* __restrict__ c1)
{
    const int i = blockIdx.x * 256 + threadIdx.x;   // B*H threads
    const bf16 vb = (bf16)ctx[i];
    h0[i] = vb; h1[i] = vb; c0[i] = 0.0f; c1[i] = 0.0f;
}

// ---------------------------------------------------------------------------
// Per-step pointwise / attention / argmax
// ---------------------------------------------------------------------------
__global__ void lstm_pw_k(const float* __restrict__ gates, float* __restrict__ c,
                          bf16* __restrict__ hb, bf16* __restrict__ xcat, int writeXcat)
{
    const int idx = blockIdx.x * 256 + threadIdx.x;   // B*H threads
    const int b = idx >> 10, j = idx & (H_ - 1);
    const float* g = gates + (long)b * G_;
    const float i_ = sigm_(g[j]);
    const float f_ = sigm_(g[H_ + j]);
    const float gg = tanhf(g[2 * H_ + j]);
    const float o_ = sigm_(g[3 * H_ + j]);
    const float cn = f_ * c[idx] + i_ * gg;
    c[idx] = cn;
    const float h = o_ * tanhf(cn);
    hb[idx] = (bf16)h;
    if (writeXcat) xcat[(long)b * K2_ + H_ + j] = (bf16)h;
}

__global__ __launch_bounds__(256) void attention_k(
    const bf16* __restrict__ qb, const bf16* __restrict__ keys,
    const bf16* __restrict__ vals, const float* __restrict__ mask,
    float* __restrict__ ctx, bf16* __restrict__ xcat)
{
    const int b = blockIdx.x, tid = threadIdx.x;
    __shared__ float qs[H_];
    __shared__ float sc[NREG];
    for (int e = tid; e < H_; e += 256) qs[e] = (float)qb[(long)b * H_ + e];
    __syncthreads();
    if (tid < NREG) {
        const bf16* kr = keys + ((long)b * NREG + tid) * H_;
        float s = 0.0f;
        for (int e = 0; e < H_; ++e) s += (float)kr[e] * qs[e];
        sc[tid] = s + mask[b * NREG + tid];
    }
    __syncthreads();
    if (tid == 0) {
        float mx = -INFINITY;
        for (int n = 0; n < NREG; ++n) mx = fmaxf(mx, sc[n]);
        float sum = 0.0f;
        for (int n = 0; n < NREG; ++n) { sc[n] = __expf(sc[n] - mx); sum += sc[n]; }
        const float inv = 1.0f / sum;
        for (int n = 0; n < NREG; ++n) sc[n] *= inv;
    }
    __syncthreads();
    for (int e = tid; e < H_; e += 256) {
        float a = 0.0f;
        for (int n = 0; n < NREG; ++n) a += sc[n] * (float)vals[((long)b * NREG + n) * H_ + e];
        ctx[(long)b * H_ + e] = a;
        xcat[(long)b * K2_ + e] = (bf16)a;
    }
}

__global__ __launch_bounds__(256) void argmax_embed_k(
    const float* __restrict__ logits, const float* __restrict__ embed, bf16* __restrict__ xb)
{
    const int b = blockIdx.x, tid = threadIdx.x;
    __shared__ float bv[256];
    __shared__ int   bi[256];
    float best = -INFINITY; int bidx = 0x7fffffff;
    for (int n = tid; n < V_; n += 256) {
        const float v = logits[(long)b * V_ + n];
        if (v > best || (v == best && n < bidx)) { best = v; bidx = n; }
    }
    bv[tid] = best; bi[tid] = bidx; __syncthreads();
    for (int s = 128; s > 0; s >>= 1) {
        if (tid < s) {
            if (bv[tid + s] > bv[tid] || (bv[tid + s] == bv[tid] && bi[tid + s] < bi[tid])) {
                bv[tid] = bv[tid + s]; bi[tid] = bi[tid + s];
            }
        }
        __syncthreads();
    }
    const int idx = bi[0];
    for (int e = tid; e < H_; e += 256) xb[(long)b * H_ + e] = (bf16)embed[(long)idx * H_ + e];
}

// ---------------------------------------------------------------------------
// Host driver
// ---------------------------------------------------------------------------
extern "C" void kernel_launch(void* const* d_in, const int* in_sizes, int n_in,
                              void* d_out, int out_size, void* d_ws, size_t ws_size,
                              hipStream_t stream) {
    (void)in_sizes; (void)n_in; (void)out_size; (void)ws_size;
    const float* roi_features = (const float*)d_in[0];
    const float* img_feat     = (const float*)d_in[1];
    const float* projcnn_w    = (const float*)d_in[2];
    const float* projcnn_b    = (const float*)d_in[3];
    const float* q_w = (const float*)d_in[4];  const float* q_b = (const float*)d_in[5];
    const float* k_w = (const float*)d_in[6];  const float* k_b = (const float*)d_in[7];
    const float* v_w = (const float*)d_in[8];  const float* v_b = (const float*)d_in[9];
    const float* W_ih = (const float*)d_in[10]; const float* W_hh = (const float*)d_in[11];
    const float* b_ih = (const float*)d_in[12]; const float* b_hh = (const float*)d_in[13];
    const float* proj_w = (const float*)d_in[14]; const float* proj_b = (const float*)d_in[15];
    const float* embed  = (const float*)d_in[16];
    float* out = (float*)d_out;

    // workspace bump allocator (1KB aligned so packed-B fragments stay aligned)
    char* w = (char*)d_ws; size_t off = 0;
    auto alloc = [&](size_t bytes) -> void* {
        void* p = w + off; off = (off + bytes + 1023) & ~(size_t)1023; return p;
    };
    bf16* qT     = (bf16*)alloc((size_t)H_ * H_ * 2);      // packed B
    bf16* kT     = (bf16*)alloc((size_t)H_ * H_ * 2);      // packed B
    bf16* vT     = (bf16*)alloc((size_t)H_ * H_ * 2);      // packed B
    bf16* WihT   = (bf16*)alloc((size_t)2 * H_ * G_ * 2);  // packed B
    bf16* WhhT   = (bf16*)alloc((size_t)2 * H_ * G_ * 2);  // packed B
    bf16* projT  = (bf16*)alloc((size_t)K2_ * VP_ * 2);    // packed B
    float* bsum  = (float*)alloc((size_t)2 * G_ * 4);
    bf16* roi_b  = (bf16*)alloc((size_t)B_ * NREG * H_ * 2);
    bf16* keysb  = (bf16*)alloc((size_t)B_ * NREG * H_ * 2);
    bf16* valsb  = (bf16*)alloc((size_t)B_ * NREG * H_ * 2);
    float* mask  = (float*)alloc((size_t)B_ * NREG * 4);
    bf16* xb     = (bf16*)alloc((size_t)B_ * H_ * 2);
    bf16* hb0    = (bf16*)alloc((size_t)B_ * H_ * 2);
    bf16* hb1    = (bf16*)alloc((size_t)B_ * H_ * 2);
    float* c0    = (float*)alloc((size_t)B_ * H_ * 4);
    float* c1    = (float*)alloc((size_t)B_ * H_ * 4);
    float* gates = (float*)alloc((size_t)B_ * G_ * 4);
    bf16* qbuf   = (bf16*)alloc((size_t)B_ * H_ * 2);
    float* ctx   = (float*)alloc((size_t)B_ * H_ * 4);
    bf16* xcat   = (bf16*)alloc((size_t)B_ * K2_ * 2);
    float* lgts  = (float*)alloc((size_t)B_ * V_ * 4);

    auto cdiv = [](long a, long b) { return (int)((a + b - 1) / b); };

    // ---- one-time: weight convert to fragment-packed bf16 ----
    transpose_pack_k<<<cdiv((long)H_ * H_, 256), 256, 0, stream>>>(q_w, qT, H_, H_, H_);
    transpose_pack_k<<<cdiv((long)H_ * H_, 256), 256, 0, stream>>>(k_w, kT, H_, H_, H_);
    transpose_pack_k<<<cdiv((long)H_ * H_, 256), 256, 0, stream>>>(v_w, vT, H_, H_, H_);
    for (int l = 0; l < 2; ++l) {
        transpose_pack_k<<<cdiv((long)H_ * G_, 256), 256, 0, stream>>>(
            W_ih + (long)l * G_ * H_, WihT + (long)l * H_ * G_, G_, H_, G_);
        transpose_pack_k<<<cdiv((long)H_ * G_, 256), 256, 0, stream>>>(
            W_hh + (long)l * G_ * H_, WhhT + (long)l * H_ * G_, G_, H_, G_);
    }
    transpose_pack_k<<<cdiv((long)K2_ * VP_, 256), 256, 0, stream>>>(proj_w, projT, V_, K2_, VP_);
    add_bias_k<<<cdiv(2 * G_, 256), 256, 0, stream>>>(b_ih, b_hh, bsum, 2 * G_);

    // ---- one-time: build roi (bf16) + mask, then keys/vals via WMMA+tanh ----
    proj_img_k<<<B_, 256, 0, stream>>>(img_feat, projcnn_w, projcnn_b, roi_b, mask);
    prep_roi_k<<<B_ * R_, 256, 0, stream>>>(roi_features, roi_b, mask);
    wmma_gemm_k<1><<<dim3(H_ / 64, (B_ * NREG) / 128), 128, 0, stream>>>(
        roi_b, kT, nullptr, nullptr, k_b, nullptr, keysb, nullptr,
        B_ * NREG, H_, H_, H_, H_);
    wmma_gemm_k<1><<<dim3(H_ / 64, (B_ * NREG) / 128), 128, 0, stream>>>(
        roi_b, vT, nullptr, nullptr, v_b, nullptr, valsb, nullptr,
        B_ * NREG, H_, H_, H_, H_);

    const dim3 gGates(G_ / 64, 1);                   // (64, 1) : 8 M-tiles per wave
    const dim3 gQ(H_ / 64, 1);                       // (16, 1)
    const dim3 gLogits(cdiv(VP_ / 16, 4), 1);        // (157, 1)

    // ---- t = 0 : start token path ----
    set_start_k<<<B_, 256, 0, stream>>>(embed, xb, xcat);
    wmma_gemm_k<1><<<gQ, 128, 0, stream>>>(xb, qT, nullptr, nullptr, q_b,
                                           nullptr, qbuf, nullptr, B_, H_, H_, H_, H_);
    attention_k<<<B_, 256, 0, stream>>>(qbuf, keysb, valsb, mask, ctx, xcat);
    init_state_k<<<(B_ * H_) / 256, 256, 0, stream>>>(ctx, hb0, hb1, c0, c1);
    wmma_gemm_k<2><<<gLogits, 128, 0, stream>>>(xcat, projT, nullptr, nullptr, proj_b,
                                                lgts, nullptr, out + 0,
                                                B_, VP_, K2_, K2_, 0);
    argmax_embed_k<<<B_, 256, 0, stream>>>(lgts, embed, xb);

    // ---- t = 1 .. T-1 : LSTM -> attention -> logits -> argmax feedback ----
    for (int t = 1; t < T_; ++t) {
        for (int l = 0; l < 2; ++l) {
            const bf16* A0 = (l == 0) ? xb  : hb0;   // layer input
            const bf16* A1 = (l == 0) ? hb0 : hb1;   // previous hidden
            wmma_gemm_k<0><<<gGates, 128, 0, stream>>>(
                A0, WihT + (long)l * H_ * G_, A1, WhhT + (long)l * H_ * G_,
                bsum + (long)l * G_, gates, nullptr, nullptr,
                B_, G_, H_, H_, G_);
            lstm_pw_k<<<(B_ * H_) / 256, 256, 0, stream>>>(
                gates, (l == 0) ? c0 : c1, (l == 0) ? hb0 : hb1, xcat, l == 1);
        }
        wmma_gemm_k<1><<<gQ, 128, 0, stream>>>(hb1, qT, nullptr, nullptr, q_b,
                                               nullptr, qbuf, nullptr, B_, H_, H_, H_, H_);
        attention_k<<<B_, 256, 0, stream>>>(qbuf, keysb, valsb, mask, ctx, xcat);
        wmma_gemm_k<2><<<gLogits, 128, 0, stream>>>(xcat, projT, nullptr, nullptr, proj_b,
                                                    lgts, nullptr, out + t,
                                                    B_, VP_, K2_, K2_, 0);
        argmax_embed_k<<<B_, 256, 0, stream>>>(lgts, embed, xb);
    }
}
